// TiMAEMambaDecoder_85263690760219
// MI455X (gfx1250) — matrix operand order repaired and checked by reading, not compiled, not run on hardware
//
#include <hip/hip_runtime.h>
#include <hip/hip_fp16.h>

typedef __attribute__((ext_vector_type(16))) _Float16 v16h;
typedef __attribute__((ext_vector_type(8)))  float    v8f;

#define Bn        8
#define DEC_LEN   129
#define ENC_LEN   512
#define ENC_EMB   512
#define EMB       768
#define OUTD      768
#define D_INNER   1536
#define D_STATE   64
#define HEADDIM   64
#define NHEADS    24
#define DCONV     4
#define CONV_DIM  1664
#define D_IN_PROJ 3224
#define L_TOT     513
#define ROWS      (Bn * L_TOT)   // 4104

// gfx1250 async global->LDS path (ASYNCcnt), with safe fallback
#if defined(__AMDGCN__) && defined(__gfx1250__) && \
    __has_builtin(__builtin_amdgcn_global_load_async_to_lds_b128) && \
    __has_builtin(__builtin_amdgcn_s_wait_asynccnt)
#define USE_ASYNC 1
#else
#define USE_ASYNC 0
#endif

// exact parameter type per hipcc diagnostic: int __vector(4) in AS1 (__device__) / AS3 (__shared__)
typedef int v4i_ __attribute__((vector_size(16)));
#define GLOBAL_V4(p) ((__attribute__((address_space(1))) v4i_*)(p))
#define LDS_V4(p)    ((__attribute__((address_space(3))) v4i_*)(p))

__device__ __forceinline__ float siluf(float x)     { return x / (1.f + __expf(-x)); }
__device__ __forceinline__ float softplusf(float x) { return (x > 20.f) ? x : log1pf(__expf(x)); }

// ---------------------------------------------------------------- converts
__global__ void f32_to_f16_kernel(const float* __restrict__ s, _Float16* __restrict__ d, size_t n) {
    size_t i = (size_t)blockIdx.x * blockDim.x + threadIdx.x;
    if (i < n) d[i] = (_Float16)s[i];
}

// ---------------------------------------------------------------- WMMA GEMM
// C[M,N] = A[M,K](f16 rowmajor) * W[N,K]^T (f16 weight rows = output cols) + bias
// block = 256 threads (8 waves); tile = 64(M) x 128(N); wave sub-tile 32x32 (4 wmma);
// K-step 32; double-buffered LDS fed by async global->LDS DMA when available.
__global__ __launch_bounds__(256)
void wmma_gemm_f16(const _Float16* __restrict__ A, const _Float16* __restrict__ W,
                   const float* __restrict__ bias, float* __restrict__ C,
                   int M, int N, int K)
{
    __shared__ _Float16 As[2][64 * 32];
    __shared__ _Float16 Bs[2][128 * 32];

    const int t      = threadIdx.x;
    const int lane   = t & 31;
    const int wave   = t >> 5;
    const int wave_m = wave & 1;          // 0..1  -> M offset 0/32
    const int wave_n = wave >> 1;         // 0..3  -> N offset 0/32/64/96
    const int m0     = blockIdx.x * 64;
    const int n0b    = blockIdx.y * 128;

    // fragment gather indices (ISA 7.12.2, wave32, 16-bit A 16x32 / B 32x16)
    const int am   = lane & 15;
    const int aklo = (lane < 16) ? 0 : 8;   // A: halves 0..7 -> K=aklo+h ; 8..15 -> K=aklo+16+(h-8)
    const int bk0  = (lane < 16) ? 0 : 8;   // B: halves 0..7 -> K=bk0+h ; 8..15 -> K=bk0+16+(h-8)
    const int bn   = lane & 15;

    v8f acc[2][2] = {};

    // ---- register-path staging (zero-filled edges) ----
    auto stage_reg = [&](_Float16* Asb, _Float16* Bsb, int k0) {
        {
            int row = t >> 2, ch = t & 3;           // A: 64 rows x 4 chunks of 8 halves
            int gm  = m0 + row;
            uint4 v = make_uint4(0u, 0u, 0u, 0u);
            if (gm < M) {
                v = *(const uint4*)(A + (size_t)gm * K + k0 + ch * 8);
                if (k0 + 32 < K) __builtin_prefetch(A + (size_t)gm * K + k0 + 32 + ch * 8, 0, 0);
            }
            *(uint4*)(Asb + row * 32 + ch * 8) = v;
        }
        {
            int row = t >> 1, ho = (t & 1) * 16;    // B: 128 rows x 2 chunks of 16 halves
            int gn  = n0b + row;
            uint4 v0 = make_uint4(0u, 0u, 0u, 0u), v1 = v0;
            if (gn < N) {
                const uint4* p = (const uint4*)(W + (size_t)gn * K + k0 + ho);
                v0 = p[0];
                v1 = p[1];
                if (k0 + 32 < K) __builtin_prefetch(W + (size_t)gn * K + k0 + 32 + ho, 0, 0);
            }
            *(uint4*)(Bsb + row * 32 + ho)     = v0;
            *(uint4*)(Bsb + row * 32 + ho + 8) = v1;
        }
    };

    // ---- fragment gather + 4 WMMAs ----
    auto compute = [&](const _Float16* Asb, const _Float16* Bsb) {
        v16h a0, a1, b0, b1;
        const _Float16* ar0 = Asb + (wave_m * 32 + am) * 32;
        const _Float16* ar1 = Asb + (wave_m * 32 + 16 + am) * 32;
        #pragma unroll
        for (int h = 0; h < 8; ++h) { a0[h] = ar0[aklo + h];          a1[h] = ar1[aklo + h]; }
        #pragma unroll
        for (int h = 0; h < 8; ++h) { a0[8 + h] = ar0[aklo + 16 + h]; a1[8 + h] = ar1[aklo + 16 + h]; }
        const _Float16* br0 = Bsb + (wave_n * 32 + bn) * 32;
        const _Float16* br1 = Bsb + (wave_n * 32 + 16 + bn) * 32;
        #pragma unroll
        for (int h = 0; h < 8; ++h) { b0[h] = br0[bk0 + h]; b1[h] = br1[bk0 + h]; }
        #pragma unroll
        for (int h = 0; h < 8; ++h) { b0[8 + h] = br0[bk0 + 16 + h]; b1[8 + h] = br1[bk0 + 16 + h]; }
        acc[0][0] = __builtin_amdgcn_wmma_f32_16x16x32_f16(false, a0, false, b0, (short)0, acc[0][0], false, false);
        acc[0][1] = __builtin_amdgcn_wmma_f32_16x16x32_f16(false, a0, false, b1, (short)0, acc[0][1], false, false);
        acc[1][0] = __builtin_amdgcn_wmma_f32_16x16x32_f16(false, a1, false, b0, (short)0, acc[1][0], false, false);
        acc[1][1] = __builtin_amdgcn_wmma_f32_16x16x32_f16(false, a1, false, b1, (short)0, acc[1][1], false, false);
    };

#if USE_ASYNC
    auto stage_async = [&](_Float16* Asb, _Float16* Bsb, int k0) {
        {
            int row = t >> 2, ch = t & 3;
            __builtin_amdgcn_global_load_async_to_lds_b128(
                GLOBAL_V4(A + (size_t)(m0 + row) * K + k0 + ch * 8),
                LDS_V4(Asb + row * 32 + ch * 8), 0, 0);
        }
        {
            int row = t >> 1, ho = (t & 1) * 16;
            const _Float16* src = W + (size_t)(n0b + row) * K + k0 + ho;
            _Float16*       dst = Bsb + row * 32 + ho;
            __builtin_amdgcn_global_load_async_to_lds_b128(
                GLOBAL_V4(src), LDS_V4(dst), 0, 0);
            __builtin_amdgcn_global_load_async_to_lds_b128(
                GLOBAL_V4(src + 8), LDS_V4(dst + 8), 0, 0);
        }
    };
    const bool full_tile = (m0 + 64 <= M) && (n0b + 128 <= N);  // block-uniform
    if (full_tile) {
        const int nk = K / 32;
        stage_async(As[0], Bs[0], 0);
        for (int i = 0; i < nk; ++i) {
            __builtin_amdgcn_s_wait_asynccnt(0);
            __syncthreads();                                  // all waves' DMA landed
            if (i + 1 < nk) stage_async(As[(i + 1) & 1], Bs[(i + 1) & 1], (i + 1) * 32);
            compute(As[i & 1], Bs[i & 1]);
        }
    } else
#endif
    {
        for (int k0 = 0; k0 < K; k0 += 32) {
            stage_reg(As[0], Bs[0], k0);
            __syncthreads();
            compute(As[0], Bs[0]);
            __syncthreads();
        }
    }

    // epilogue (ISA 32-bit C/D 16x16 layout)
    #pragma unroll
    for (int mt = 0; mt < 2; ++mt) {
        #pragma unroll
        for (int nt = 0; nt < 2; ++nt) {
            int n = n0b + wave_n * 32 + nt * 16 + bn;
            float bv = (bias != nullptr && n < N) ? bias[n] : 0.f;
            #pragma unroll
            for (int r = 0; r < 8; ++r) {
                int m = m0 + wave_m * 32 + mt * 16 + r + ((lane < 16) ? 0 : 8);
                if (m < M && n < N) C[(size_t)m * N + n] = acc[mt][nt][r] + bv;
            }
        }
    }
}

// ---------------------------------------------------------------- unshuffle
__global__ void gather_embed_kernel(const float* __restrict__ xe, const int* __restrict__ ids,
                                    const float* __restrict__ mask_tok, _Float16* __restrict__ xin)
{
    int row = blockIdx.x;                 // 0..ROWS-1
    int b = row / L_TOT, j = row % L_TOT;
    int src;
    if (j == ENC_LEN) src = DEC_LEN - 1;  // cls token appended at end
    else {
        int s = ids[b * ENC_LEN + j];
        src = (s < DEC_LEN - 1) ? s : -1; // -1 => mask token
    }
    for (int e = threadIdx.x; e < EMB; e += blockDim.x) {
        float v = (src >= 0) ? xe[((size_t)b * DEC_LEN + src) * EMB + e] : mask_tok[e];
        xin[(size_t)row * EMB + e] = (_Float16)v;
    }
}

// ---------------------------------------------------------------- dt / dA
__global__ void dt_kernel(const float* __restrict__ zx, const float* __restrict__ dt_bias,
                          const float* __restrict__ A_log,
                          float* __restrict__ dts, float* __restrict__ dAa)
{
    int i = blockIdx.x * blockDim.x + threadIdx.x;
    if (i >= ROWS * NHEADS) return;
    int row = i / NHEADS, h = i % NHEADS;
    float d = softplusf(zx[(size_t)row * D_IN_PROJ + D_INNER + CONV_DIM + h] + dt_bias[h]);
    dts[i] = d;
    dAa[i] = __expf(d * (-__expf(A_log[h])));
}

// ---------------------------------------------------------------- causal depthwise conv + SiLU
__global__ void conv_kernel(const float* __restrict__ zx, const float* __restrict__ cw,
                            const float* __restrict__ cb, float* __restrict__ xbc)
{
    int i = blockIdx.x * blockDim.x + threadIdx.x;
    if (i >= Bn * CONV_DIM) return;
    int b = i / CONV_DIM, c = i % CONV_DIM;
    float w0 = cw[c * 4 + 0], w1 = cw[c * 4 + 1], w2 = cw[c * 4 + 2], w3 = cw[c * 4 + 3];
    float bb = cb[c];
    float x0 = 0.f, x1 = 0.f, x2 = 0.f;
    for (int l = 0; l < L_TOT; ++l) {
        float x3 = zx[((size_t)b * L_TOT + l) * D_IN_PROJ + D_INNER + c];
        float s  = x0 * w0 + x1 * w1 + x2 * w2 + x3 * w3 + bb;
        xbc[((size_t)b * L_TOT + l) * CONV_DIM + c] = siluf(s);
        x0 = x1; x1 = x2; x2 = x3;
    }
}

// ---------------------------------------------------------------- SSM scan (state in registers)
// block = one (b, head); thread t owns p = t/4, n-slice = (t&3)*16 .. +16
__global__ __launch_bounds__(256)
void scan_kernel(const float* __restrict__ xbc, const float* __restrict__ dts,
                 const float* __restrict__ dAa, const float* __restrict__ Dvec,
                 float* __restrict__ y)
{
    int bh = blockIdx.x;
    int b = bh / NHEADS, nh = bh % NHEADS;
    int t  = threadIdx.x;
    int p  = t >> 2;
    int nq = (t & 3) * 16;
    float h[16];
    #pragma unroll
    for (int i = 0; i < 16; ++i) h[i] = 0.f;
    const float Dh = Dvec[nh];

    for (int l = 0; l < L_TOT; ++l) {
        size_t row = (size_t)b * L_TOT + l;
        float dt = dts[row * NHEADS + nh];
        float da = dAa[row * NHEADS + nh];
        const float* base = xbc + row * CONV_DIM;
        if (l + 1 < L_TOT) {  // prefetch next timestep's xs / B cachelines
            __builtin_prefetch(base + CONV_DIM + nh * HEADDIM + p, 0, 0);
            __builtin_prefetch(base + CONV_DIM + D_INNER + nq, 0, 0);
        }
        float xs  = base[nh * HEADDIM + p];
        float xdt = xs * dt;
        float accv = 0.f;
        #pragma unroll
        for (int i = 0; i < 4; ++i) {
            float4 Bv = *(const float4*)(base + D_INNER + nq + i * 4);
            float4 Cv = *(const float4*)(base + D_INNER + D_STATE + nq + i * 4);
            int k = i * 4;
            h[k + 0] = h[k + 0] * da + xdt * Bv.x; accv += h[k + 0] * Cv.x;
            h[k + 1] = h[k + 1] * da + xdt * Bv.y; accv += h[k + 1] * Cv.y;
            h[k + 2] = h[k + 2] * da + xdt * Bv.z; accv += h[k + 2] * Cv.z;
            h[k + 3] = h[k + 3] * da + xdt * Bv.w; accv += h[k + 3] * Cv.w;
        }
        accv += __shfl_xor(accv, 1, 32);
        accv += __shfl_xor(accv, 2, 32);
        if ((t & 3) == 0)
            y[row * D_INNER + nh * HEADDIM + p] = accv + Dh * xs;
    }
}

// ---------------------------------------------------------------- gated RMSNorm -> f16
__global__ __launch_bounds__(256)
void gated_rms_kernel(const float* __restrict__ y, const float* __restrict__ zx,
                      const float* __restrict__ rms_w, _Float16* __restrict__ y16)
{
    __shared__ float red[256];
    int row = blockIdx.x, t = threadIdx.x;
    float g[6];
    float ss = 0.f;
    #pragma unroll
    for (int i = 0; i < 6; ++i) {
        int e = t + i * 256;
        float gv = y[(size_t)row * D_INNER + e] * siluf(zx[(size_t)row * D_IN_PROJ + e]);
        g[i] = gv;
        ss += gv * gv;
    }
    red[t] = ss; __syncthreads();
    for (int s = 128; s > 0; s >>= 1) { if (t < s) red[t] += red[t + s]; __syncthreads(); }
    float r = rsqrtf(red[0] / (float)D_INNER + 1e-5f);
    #pragma unroll
    for (int i = 0; i < 6; ++i) {
        int e = t + i * 256;
        y16[(size_t)row * D_INNER + e] = (_Float16)(g[i] * r * rms_w[e]);
    }
}

// ---------------------------------------------------------------- LayerNorm -> f16
__global__ __launch_bounds__(256)
void layernorm_kernel(const float* __restrict__ x, const float* __restrict__ w,
                      const float* __restrict__ bta, _Float16* __restrict__ o)
{
    __shared__ float red[256];
    int row = blockIdx.x, t = threadIdx.x;
    float v[3];
    float s = 0.f;
    #pragma unroll
    for (int i = 0; i < 3; ++i) { v[i] = x[(size_t)row * EMB + t + i * 256]; s += v[i]; }
    red[t] = s; __syncthreads();
    for (int st = 128; st > 0; st >>= 1) { if (t < st) red[t] += red[t + st]; __syncthreads(); }
    float mu = red[0] / (float)EMB;
    __syncthreads();
    float s2 = 0.f;
    #pragma unroll
    for (int i = 0; i < 3; ++i) { float d = v[i] - mu; s2 += d * d; }
    red[t] = s2; __syncthreads();
    for (int st = 128; st > 0; st >>= 1) { if (t < st) red[t] += red[t + st]; __syncthreads(); }
    float r = rsqrtf(red[0] / (float)EMB + 1e-5f);
    #pragma unroll
    for (int i = 0; i < 3; ++i) {
        int e = t + i * 256;
        o[(size_t)row * EMB + e] = (_Float16)((v[i] - mu) * r * w[e] + bta[e]);
    }
}

// ---------------------------------------------------------------- strip cls row -> d_out
__global__ void copy_out_kernel(const float* __restrict__ src, float* __restrict__ dst)
{
    size_t i = (size_t)blockIdx.x * blockDim.x + threadIdx.x;
    size_t n = (size_t)Bn * ENC_LEN * OUTD;
    if (i >= n) return;
    size_t e = i % OUTD;
    size_t r = i / OUTD;
    size_t l = r % ENC_LEN, b = r / ENC_LEN;
    dst[i] = src[((size_t)b * L_TOT + l) * OUTD + e];
}

// =====================================================================
extern "C" void kernel_launch(void* const* d_in, const int* in_sizes, int n_in,
                              void* d_out, int out_size, void* d_ws, size_t ws_size,
                              hipStream_t stream)
{
    const float* x         = (const float*)d_in[0];
    const int*   ids       = (const int*)  d_in[1];
    const float* mask_tok  = (const float*)d_in[2];
    const float* emb_w     = (const float*)d_in[3];
    const float* emb_b     = (const float*)d_in[4];
    const float* in_w      = (const float*)d_in[5];
    const float* conv_w    = (const float*)d_in[6];
    const float* conv_b    = (const float*)d_in[7];
    const float* dt_bias   = (const float*)d_in[8];
    const float* A_log     = (const float*)d_in[9];
    const float* Dvec      = (const float*)d_in[10];
    const float* rms_w     = (const float*)d_in[11];
    const float* out_w     = (const float*)d_in[12];
    const float* ln_w      = (const float*)d_in[13];
    const float* ln_b      = (const float*)d_in[14];
    const float* proj_w    = (const float*)d_in[15];
    const float* proj_b    = (const float*)d_in[16];

    char* base = (char*)d_ws;
    size_t off = 0;
    auto alloc = [&](size_t bytes) -> char* {
        off = (off + 255) & ~(size_t)255;
        char* p = base + off;
        off += bytes;
        return p;
    };

    const size_t XROWS = (size_t)Bn * DEC_LEN;   // 1032
    _Float16* xf16    = (_Float16*)alloc(XROWS * ENC_EMB * 2);
    _Float16* embw16  = (_Float16*)alloc((size_t)EMB * ENC_EMB * 2);
    _Float16* inw16   = (_Float16*)alloc((size_t)D_IN_PROJ * EMB * 2);
    _Float16* outw16  = (_Float16*)alloc((size_t)EMB * D_INNER * 2);
    _Float16* projw16 = (_Float16*)alloc((size_t)OUTD * EMB * 2);
    float*    xemb    = (float*)   alloc(XROWS * EMB * 4);
    _Float16* xin16   = (_Float16*)alloc((size_t)ROWS * EMB * 2);
    float*    zx      = (float*)   alloc((size_t)ROWS * D_IN_PROJ * 4);
    float*    dts     = (float*)   alloc((size_t)ROWS * NHEADS * 4);
    float*    dAa     = (float*)   alloc((size_t)ROWS * NHEADS * 4);
    float*    xbc     = (float*)   alloc((size_t)ROWS * CONV_DIM * 4);
    float*    yb      = (float*)   alloc((size_t)ROWS * D_INNER * 4);
    _Float16* y16     = (_Float16*)alloc((size_t)ROWS * D_INNER * 2);
    float*    mo      = (float*)   alloc((size_t)ROWS * EMB * 4);
    _Float16* ln16    = (_Float16*)alloc((size_t)ROWS * EMB * 2);
    float*    po      = (float*)   alloc((size_t)ROWS * OUTD * 4);
    (void)ws_size; (void)n_in; (void)in_sizes; (void)out_size;

    auto cvt = [&](const float* s, _Float16* d, size_t n) {
        f32_to_f16_kernel<<<dim3((unsigned)((n + 255) / 256)), dim3(256), 0, stream>>>(s, d, n);
    };
    cvt(x,      xf16,    XROWS * ENC_EMB);
    cvt(emb_w,  embw16,  (size_t)EMB * ENC_EMB);
    cvt(in_w,   inw16,   (size_t)D_IN_PROJ * EMB);
    cvt(out_w,  outw16,  (size_t)EMB * D_INNER);
    cvt(proj_w, projw16, (size_t)OUTD * EMB);

    // 1) embed GEMM: (1032 x 512) x (768 x 512)^T + emb_b
    wmma_gemm_f16<<<dim3((XROWS + 63) / 64, (EMB + 127) / 128), dim3(256), 0, stream>>>(
        xf16, embw16, emb_b, xemb, (int)XROWS, EMB, ENC_EMB);

    // 2) unshuffle + cls append -> f16 activations
    gather_embed_kernel<<<dim3(ROWS), dim3(256), 0, stream>>>(xemb, ids, mask_tok, xin16);

    // 3) in_proj GEMM: (4104 x 768) x (3224 x 768)^T
    wmma_gemm_f16<<<dim3((ROWS + 63) / 64, (D_IN_PROJ + 127) / 128), dim3(256), 0, stream>>>(
        xin16, inw16, nullptr, zx, ROWS, D_IN_PROJ, EMB);

    // 4) dt softplus + dA
    {
        int n = ROWS * NHEADS;
        dt_kernel<<<dim3((n + 255) / 256), dim3(256), 0, stream>>>(zx, dt_bias, A_log, dts, dAa);
    }

    // 5) depthwise causal conv + SiLU
    {
        int n = Bn * CONV_DIM;
        conv_kernel<<<dim3((n + 255) / 256), dim3(256), 0, stream>>>(zx, conv_w, conv_b, xbc);
    }

    // 6) SSM scan
    scan_kernel<<<dim3(Bn * NHEADS), dim3(256), 0, stream>>>(xbc, dts, dAa, Dvec, yb);

    // 7) gated RMSNorm -> f16
    gated_rms_kernel<<<dim3(ROWS), dim3(256), 0, stream>>>(yb, zx, rms_w, y16);

    // 8) out_proj GEMM: (4104 x 1536) x (768 x 1536)^T
    wmma_gemm_f16<<<dim3((ROWS + 63) / 64, (EMB + 127) / 128), dim3(256), 0, stream>>>(
        y16, outw16, nullptr, mo, ROWS, EMB, D_INNER);

    // 9) LayerNorm -> f16
    layernorm_kernel<<<dim3(ROWS), dim3(256), 0, stream>>>(mo, ln_w, ln_b, ln16);

    // 10) final proj GEMM + bias
    wmma_gemm_f16<<<dim3((ROWS + 63) / 64, (OUTD + 127) / 128), dim3(256), 0, stream>>>(
        ln16, projw16, proj_b, po, ROWS, OUTD, EMB);

    // 11) drop cls row -> d_out (B, 512, 768) f32
    {
        size_t n = (size_t)Bn * ENC_LEN * OUTD;
        copy_out_kernel<<<dim3((unsigned)((n + 255) / 256)), dim3(256), 0, stream>>>(po, (float*)d_out);
    }
}
